// Net_71760313582335
// MI455X (gfx1250) — matrix operand order repaired and checked by reading, not compiled, not run on hardware
//
#include <hip/hip_runtime.h>
#include <hip/hip_bf16.h>

// CDNA5 / gfx1250 (wave32) GCN forward.
// Workspace layout (floats): [dinv: N][xw1: N*16][agg1: N*16][hw2: N*10][agg2: N*10]

typedef __attribute__((ext_vector_type(16))) __bf16 v16bf;
typedef __attribute__((ext_vector_type(8)))  float  v8f;

#define FIN  128
#define HID  16
#define NCLS 10

// ---------------- degree / normalization ----------------

__global__ void k_deg_init(float* __restrict__ deg, int n) {
  int i = blockIdx.x * blockDim.x + threadIdx.x;
  if (i < n) deg[i] = 1.0f;  // self loop contributes 1
}

__global__ void k_deg_scatter(const long long* __restrict__ col,
                              float* __restrict__ deg, int E) {
  int e = blockIdx.x * blockDim.x + threadIdx.x;
  if (e < E) {
    __builtin_prefetch(col + e + 4096, 0, 0);   // global_prefetch_b8
    atomicAdd(&deg[col[e]], 1.0f);
  }
}

__global__ void k_dinv(float* __restrict__ deg, int n) {
  int i = blockIdx.x * blockDim.x + threadIdx.x;
  if (i < n) deg[i] = rsqrtf(deg[i]);           // deg >= 1 always
}

// ---------------- layer 1: xw1 = X (N x 128) * W1 (128 x 16), WMMA bf16 ----------------
// One wave per 16-row tile; K=128 as four v_wmma_f32_16x16x32_bf16.

__global__ __launch_bounds__(256) void k_xw1(const float* __restrict__ x,
                                             const float* __restrict__ W1,
                                             float* __restrict__ xw,
                                             int n, int nTiles) {
  const int wave = threadIdx.x >> 5;
  const int lane = threadIdx.x & 31;
  const int tile = blockIdx.x * 8 + wave;
  if (tile >= nTiles) return;                 // wave-uniform: EXEC stays all-1s
  const int halfLane = lane & 15;
  const int hi       = lane >> 4;
  int row = tile * 16 + halfLane;
  if (row >= n) row = n - 1;                  // clamp loads; stores are guarded
  const float* __restrict__ xr = x + (size_t)row * FIN;

  v8f acc = {};
#pragma unroll
  for (int kc = 0; kc < FIN / 32; ++kc) {
    // A fragment 16x32 (ISA 7.12.2): lane holds row=halfLane;
    // VGPR v<4: K = hi*8 + 2v+{0,1}; v>=4: K = 16 + hi*8 + 2(v-4)+{0,1}.
    v16bf a;
#pragma unroll
    for (int v = 0; v < 8; ++v) {
      const int kb = (v < 4) ? (hi * 8 + 2 * v) : (16 + hi * 8 + 2 * (v - 4));
      const float2 p = *(const float2*)(xr + kc * 32 + kb);
      a[2 * v]     = (__bf16)p.x;
      a[2 * v + 1] = (__bf16)p.y;
    }
    // B fragment 32x16: lane holds col N=halfLane; element t<8 holds K = hi*8 + t, t>=8 holds K = 16 + hi*8 + (t-8).
    v16bf b;
#pragma unroll
    for (int t = 0; t < 16; ++t) {
      const int k = kc * 32 + ((t < 8) ? (hi * 8 + t) : (16 + hi * 8 + (t - 8)));
      b[t] = (__bf16)W1[k * HID + halfLane];
    }
    acc = __builtin_amdgcn_wmma_f32_16x16x32_bf16(false, a, false, b,
                                                  (short)0, acc, false, false);
  }
  // D: VGPR r -> row = hi*8 + r, col = halfLane
#pragma unroll
  for (int r = 0; r < 8; ++r) {
    const int m = tile * 16 + hi * 8 + r;
    if (m < n) xw[(size_t)m * HID + halfLane] = acc[r];
  }
}

// ---------------- self-loop init + edge scatter, layer 1 ----------------

__global__ void k_init_agg1(const float* __restrict__ dinv,
                            const float* __restrict__ xw,
                            float* __restrict__ agg, int total) {
  int i = blockIdx.x * blockDim.x + threadIdx.x;
  if (i < total) {
    const float d = dinv[i / HID];
    agg[i] = d * d * xw[i];
  }
}

__global__ void k_scatter1(const long long* __restrict__ rowI,
                           const long long* __restrict__ colI,
                           const float* __restrict__ dinv,
                           const float* __restrict__ xw,
                           float* __restrict__ agg, int E) {
  int e = blockIdx.x * blockDim.x + threadIdx.x;
  if (e >= E) return;
  __builtin_prefetch(rowI + e + 4096, 0, 0);
  __builtin_prefetch(colI + e + 4096, 0, 0);
  const long long r = rowI[e];
  const long long c = colI[e];
  const float nrm = dinv[r] * dinv[c];
  const float4* __restrict__ src = (const float4*)(xw + (size_t)r * HID);
  float* __restrict__ dst = agg + (size_t)c * HID;
#pragma unroll
  for (int q = 0; q < 4; ++q) {
    const float4 v = src[q];
    atomicAdd(dst + 4 * q + 0, nrm * v.x);
    atomicAdd(dst + 4 * q + 1, nrm * v.y);
    atomicAdd(dst + 4 * q + 2, nrm * v.z);
    atomicAdd(dst + 4 * q + 3, nrm * v.w);
  }
}

// ---------------- layer 2: hw2 = relu(agg1 + b1) * W2 (16 x 10), fused, single WMMA ----------------

__global__ __launch_bounds__(256) void k_mm2(const float* __restrict__ agg1,
                                             const float* __restrict__ b1,
                                             const float* __restrict__ W2,
                                             float* __restrict__ hw2,
                                             int n, int nTiles) {
  const int wave = threadIdx.x >> 5;
  const int lane = threadIdx.x & 31;
  const int tile = blockIdx.x * 8 + wave;
  if (tile >= nTiles) return;                 // wave-uniform
  const int halfLane = lane & 15;
  const int hi       = lane >> 4;
  int row = tile * 16 + halfLane;
  if (row >= n) row = n - 1;

  // A fragment: K<16 = relu(agg1+b1), K>=16 zero-padded.
  v16bf a;
#pragma unroll
  for (int v = 0; v < 8; ++v) {
#pragma unroll
    for (int j = 0; j < 2; ++j) {
      const int k = (v < 4) ? (hi * 8 + 2 * v + j) : (16 + hi * 8 + 2 * (v - 4) + j);
      float val = 0.0f;
      if (k < HID) {
        val = agg1[(size_t)row * HID + k] + b1[k];
        val = val > 0.0f ? val : 0.0f;        // ReLU
      }
      a[2 * v + j] = (__bf16)val;
    }
  }
  // B fragment: W2 16x10, zero-padded to 32x16.
  v16bf b;
#pragma unroll
  for (int t = 0; t < 16; ++t) {
    const int k = (t < 8) ? (hi * 8 + t) : (16 + hi * 8 + (t - 8));
    float w = 0.0f;
    if (k < HID && halfLane < NCLS) w = W2[k * NCLS + halfLane];
    b[t] = (__bf16)w;
  }
  v8f acc = {};
  acc = __builtin_amdgcn_wmma_f32_16x16x32_bf16(false, a, false, b,
                                                (short)0, acc, false, false);
#pragma unroll
  for (int r = 0; r < 8; ++r) {
    const int m = tile * 16 + hi * 8 + r;
    if (m < n && halfLane < NCLS) hw2[(size_t)m * NCLS + halfLane] = acc[r];
  }
}

// ---------------- self-loop init + edge scatter, layer 2 ----------------

__global__ void k_init_agg2(const float* __restrict__ dinv,
                            const float* __restrict__ hw2,
                            float* __restrict__ agg, int total) {
  int i = blockIdx.x * blockDim.x + threadIdx.x;
  if (i < total) {
    const float d = dinv[i / NCLS];
    agg[i] = d * d * hw2[i];
  }
}

__global__ void k_scatter2(const long long* __restrict__ rowI,
                           const long long* __restrict__ colI,
                           const float* __restrict__ dinv,
                           const float* __restrict__ hw2,
                           float* __restrict__ agg, int E) {
  int e = blockIdx.x * blockDim.x + threadIdx.x;
  if (e >= E) return;
  __builtin_prefetch(rowI + e + 4096, 0, 0);
  __builtin_prefetch(colI + e + 4096, 0, 0);
  const long long r = rowI[e];
  const long long c = colI[e];
  const float nrm = dinv[r] * dinv[c];
  const float2* __restrict__ src = (const float2*)(hw2 + (size_t)r * NCLS);
  float* __restrict__ dst = agg + (size_t)c * NCLS;
#pragma unroll
  for (int q = 0; q < 5; ++q) {
    const float2 v = src[q];
    atomicAdd(dst + 2 * q + 0, nrm * v.x);
    atomicAdd(dst + 2 * q + 1, nrm * v.y);
  }
}

// ---------------- bias + log_softmax ----------------

__global__ void k_logsoftmax(const float* __restrict__ agg2,
                             const float* __restrict__ b2,
                             float* __restrict__ out, int n) {
  int i = blockIdx.x * blockDim.x + threadIdx.x;
  if (i >= n) return;
  float v[NCLS];
  float m = -3.402823466e38f;
#pragma unroll
  for (int f = 0; f < NCLS; ++f) {
    v[f] = agg2[(size_t)i * NCLS + f] + b2[f];
    m = v[f] > m ? v[f] : m;
  }
  float s = 0.0f;
#pragma unroll
  for (int f = 0; f < NCLS; ++f) s += expf(v[f] - m);
  const float ls = logf(s);
#pragma unroll
  for (int f = 0; f < NCLS; ++f) out[(size_t)i * NCLS + f] = v[f] - m - ls;
}

// ---------------- host launch ----------------

extern "C" void kernel_launch(void* const* d_in, const int* in_sizes, int n_in,
                              void* d_out, int out_size, void* d_ws, size_t ws_size,
                              hipStream_t stream) {
  const float*     x   = (const float*)d_in[0];
  const long long* ei  = (const long long*)d_in[1];   // int64 [2, E]
  const float*     W1  = (const float*)d_in[2];
  const float*     b1  = (const float*)d_in[3];
  const float*     W2  = (const float*)d_in[4];
  const float*     b2  = (const float*)d_in[5];
  float*           out = (float*)d_out;

  const int N = in_sizes[0] / FIN;
  const int E = in_sizes[1] / 2;
  const long long* rowI = ei;       // source
  const long long* colI = ei + E;   // target

  float* ws   = (float*)d_ws;       // ~21.2 MB total for N=100k
  float* dinv = ws;
  float* xw1  = dinv + N;
  float* agg1 = xw1 + (size_t)N * HID;
  float* hw2  = agg1 + (size_t)N * HID;
  float* agg2 = hw2 + (size_t)N * NCLS;

  const int T = 256;
  const int nTiles = (N + 15) / 16;
  const dim3 gN((N + T - 1) / T), gE((E + T - 1) / T);
  const dim3 gT((nTiles + 7) / 8);
  const dim3 gA1(((size_t)N * HID + T - 1) / T), gA2(((size_t)N * NCLS + T - 1) / T);

  // normalization
  k_deg_init   <<<gN, T, 0, stream>>>(dinv, N);
  k_deg_scatter<<<gE, T, 0, stream>>>(colI, dinv, E);
  k_dinv       <<<gN, T, 0, stream>>>(dinv, N);

  // layer 1
  k_xw1      <<<gT, T, 0, stream>>>(x, W1, xw1, N, nTiles);
  k_init_agg1<<<gA1, T, 0, stream>>>(dinv, xw1, agg1, (int)((size_t)N * HID));
  k_scatter1 <<<gE, T, 0, stream>>>(rowI, colI, dinv, xw1, agg1, E);

  // layer 2 (relu + bias fused into the WMMA A-fragment build)
  k_mm2      <<<gT, T, 0, stream>>>(agg1, b1, W2, hw2, N, nTiles);
  k_init_agg2<<<gA2, T, 0, stream>>>(dinv, hw2, agg2, (int)((size_t)N * NCLS));
  k_scatter2 <<<gE, T, 0, stream>>>(rowI, colI, dinv, hw2, agg2, E);

  // output
  k_logsoftmax<<<gN, T, 0, stream>>>(agg2, b2, out, N);
}
